// MPTBlock_65257733096010
// MI455X (gfx1250) — matrix operand order repaired and checked by reading, not compiled, not run on hardware
//
#include <hip/hip_runtime.h>

// ---------------------------------------------------------------------------
// MPT block for MI455X (gfx1250, wave32, WMMA).
// Compute-bound (~450 GFLOP vs ~13us of HBM traffic at 23.3 TB/s): all GEMMs
// + attention run on v_wmma_f32_16x16x32_bf16 with fp32 accumulate.
// GEMM tiles are staged with CDNA5 async global->LDS copies
// (global_load_async_to_lds_b128, ASYNCcnt) and double-buffered LDS so the
// async DMA overlaps the WMMA pipeline; no VGPR staging, fewer waits.
// ---------------------------------------------------------------------------

typedef unsigned short u16;
typedef __bf16 bf16x16 __attribute__((ext_vector_type(16)));
typedef float  f32x8   __attribute__((ext_vector_type(8)));

union FragU { uint4 u[2]; bf16x16 v; };

__device__ __forceinline__ u16 f2bf(float f) {
  unsigned u = __builtin_bit_cast(unsigned, f);
  u += 0x7FFFu + ((u >> 16) & 1u);          // round-to-nearest-even
  return (u16)(u >> 16);
}

__device__ __forceinline__ f32x8 zero8() {
  f32x8 z;
#pragma unroll
  for (int i = 0; i < 8; ++i) z[i] = 0.f;
  return z;
}

// Low 32 bits of a flat LDS pointer == byte offset within the WG's LDS.
__device__ __forceinline__ unsigned lds_off(const void* p) {
  return (unsigned)(size_t)p;
}

// One wave-wide async copy: 16 bytes per lane, global -> LDS (ASYNCcnt).
__device__ __forceinline__ void async_b128(const u16* g, unsigned loff) {
  asm volatile("global_load_async_to_lds_b128 %0, %1, off"
               :: "v"(loff), "v"((unsigned long long)(size_t)g)
               : "memory");
}

__device__ __forceinline__ void wait_async0() {
  asm volatile("s_wait_asynccnt 0x0" ::: "memory");
}

// ---------------------------------------------------------------------------
// fp32 [R][C]  ->  bf16 [C][R]   (weight convert + transpose)
// ---------------------------------------------------------------------------
__global__ __launch_bounds__(256) void transpose_f32_bf16(
    const float* __restrict__ in, u16* __restrict__ out, int R, int C) {
  __shared__ float tile[32][33];
  const int bx = blockIdx.x * 32;            // col base (in)
  const int by = blockIdx.y * 32;            // row base (in)
  const int tx = threadIdx.x & 31;
  const int ty = threadIdx.x >> 5;           // 8 waves -> 4 rows each
#pragma unroll
  for (int i = 0; i < 4; ++i) {
    int r = ty * 4 + i;
    tile[r][tx] = in[(size_t)(by + r) * C + bx + tx];
  }
  __syncthreads();
#pragma unroll
  for (int i = 0; i < 4; ++i) {
    int r = ty * 4 + i;
    out[(size_t)(bx + r) * R + by + tx] = f2bf(tile[tx][r]);
  }
}

// ---------------------------------------------------------------------------
// LayerNorm row (len 2048) -> bf16
// ---------------------------------------------------------------------------
__global__ __launch_bounds__(256) void layernorm_k(
    const float* __restrict__ x, const float* __restrict__ wgt,
    u16* __restrict__ out) {
  __shared__ float red[8];
  const int row = blockIdx.x, tid = threadIdx.x;
  const int lane = tid & 31, wv = tid >> 5;
  const float* xr = x + (size_t)row * 2048;
  float v[8];
  float s = 0.f;
#pragma unroll
  for (int i = 0; i < 8; ++i) { v[i] = xr[tid + i * 256]; s += v[i]; }
#pragma unroll
  for (int m = 1; m < 32; m <<= 1) s += __shfl_xor(s, m, 32);
  if (lane == 0) red[wv] = s;
  __syncthreads();
  float tot = 0.f;
#pragma unroll
  for (int i = 0; i < 8; ++i) tot += red[i];
  const float mu = tot * (1.f / 2048.f);
  float d2 = 0.f;
#pragma unroll
  for (int i = 0; i < 8; ++i) { float d = v[i] - mu; d2 += d * d; }
#pragma unroll
  for (int m = 1; m < 32; m <<= 1) d2 += __shfl_xor(d2, m, 32);
  __syncthreads();
  if (lane == 0) red[wv] = d2;
  __syncthreads();
  float vt = 0.f;
#pragma unroll
  for (int i = 0; i < 8; ++i) vt += red[i];
  const float rs = rsqrtf(vt * (1.f / 2048.f) + 1e-5f);
#pragma unroll
  for (int i = 0; i < 8; ++i)
    out[(size_t)row * 2048 + tid + i * 256] =
        f2bf((v[i] - mu) * rs * wgt[tid + i * 256]);
}

// ---------------------------------------------------------------------------
// bf16 GEMM: C[M][N] = A[M][K] * Bt[N][K]^T, fp32 accumulate via WMMA.
// Block tile 128x128, 8 waves (4x2), wave tile 32x64 (2x4 wmma accums).
// K tile 64, double-buffered LDS filled by async global->LDS copies.
// MODE 0: split QKV -> Q[T][2048], K[T][2048], Vt[b][h][hd][seq] (bf16)
// MODE 1: outF = resid + acc (fp32)
// MODE 2: oB   = gelu_exact(acc) (bf16)
// ---------------------------------------------------------------------------
template <int MODE>
__global__ __launch_bounds__(256) void gemm_bf16(
    const u16* __restrict__ A, const u16* __restrict__ Bt,
    int M, int N, int K,
    const float* __restrict__ resid, float* __restrict__ outF,
    u16* __restrict__ o0, u16* __restrict__ o1, u16* __restrict__ o2) {
  __shared__ alignas(16) u16 sA[2][128][64];   // 32 KB
  __shared__ alignas(16) u16 sB[2][128][64];   // 32 KB
  const int tid = threadIdx.x;
  const int lane = tid & 31, w = tid >> 5;
  const int g = lane >> 4, lm = lane & 15;
  const int mBase = blockIdx.y * 128, nBase = blockIdx.x * 128;
  const int waveM = (w >> 1) * 32, waveN = (w & 1) * 64;
  const int lr = tid >> 1;            // 0..127 tile row
  const int lc = (tid & 1) * 32;      // 0 or 32 k-col (32 u16 = 64B each)

  f32x8 acc[2][4];
#pragma unroll
  for (int mt = 0; mt < 2; ++mt)
#pragma unroll
    for (int nt = 0; nt < 4; ++nt) acc[mt][nt] = zero8();

  const u16* gA = A + (size_t)(mBase + lr) * K + lc;
  const u16* gB = Bt + (size_t)(nBase + lr) * K + lc;
  const unsigned aoff0 = lds_off(&sA[0][lr][lc]), aoff1 = lds_off(&sA[1][lr][lc]);
  const unsigned boff0 = lds_off(&sB[0][lr][lc]), boff1 = lds_off(&sB[1][lr][lc]);

  // prologue: async-fill buffer 0 with tile k0=0 (8 async b128 per thread)
#pragma unroll
  for (int c = 0; c < 4; ++c) {
    async_b128(gA + c * 8, aoff0 + c * 16);
    async_b128(gB + c * 8, boff0 + c * 16);
  }

  for (int k0 = 0; k0 < K; k0 += 64) {
    const int cur = (k0 >> 6) & 1;
    wait_async0();        // my writes to buf `cur` have landed in LDS
    __syncthreads();      // everyone's writes landed; prev reads of buf cur^1 done
    if (k0 + 64 < K) {    // async-prefetch next tile into the other buffer
      const unsigned na = cur ? aoff0 : aoff1;
      const unsigned nb = cur ? boff0 : boff1;
      const u16* pa = gA + k0 + 64;
      const u16* pb = gB + k0 + 64;
#pragma unroll
      for (int c = 0; c < 4; ++c) {
        async_b128(pa + c * 8, na + c * 16);
        async_b128(pb + c * 8, nb + c * 16);
      }
    }
#pragma unroll
    for (int ks = 0; ks < 2; ++ks) {     // two 16x16x32 K-steps per tile
      FragU fa[2];
#pragma unroll
      for (int mt = 0; mt < 2; ++mt) {
        int row = waveM + mt * 16 + lm;
        fa[mt].u[0] = *(const uint4*)&sA[cur][row][ks * 32 + 16 * g];
        fa[mt].u[1] = *(const uint4*)&sA[cur][row][ks * 32 + 16 * g + 8];
      }
      FragU fb[4];
#pragma unroll
      for (int nt = 0; nt < 4; ++nt) {
        int row = waveN + nt * 16 + lm;
        fb[nt].u[0] = *(const uint4*)&sB[cur][row][ks * 32 + 16 * g];
        fb[nt].u[1] = *(const uint4*)&sB[cur][row][ks * 32 + 16 * g + 8];
      }
#pragma unroll
      for (int mt = 0; mt < 2; ++mt)
#pragma unroll
        for (int nt = 0; nt < 4; ++nt)
          acc[mt][nt] = __builtin_amdgcn_wmma_f32_16x16x32_bf16(
              false, fa[mt].v, false, fb[nt].v, (short)0, acc[mt][nt],
              false, false);
    }
  }

  // epilogue: lane holds col = lm, VGPR r holds row = r + 8g
#pragma unroll
  for (int mt = 0; mt < 2; ++mt) {
#pragma unroll
    for (int nt = 0; nt < 4; ++nt) {
#pragma unroll
      for (int r = 0; r < 8; ++r) {
        const int row = mBase + waveM + mt * 16 + 8 * g + r;
        const int col = nBase + waveN + nt * 16 + lm;
        const float v = acc[mt][nt][r];
        if constexpr (MODE == 0) {
          const int sel = col >> 11, c2 = col & 2047;
          const u16 bv = f2bf(v);
          if (sel == 0) {
            o0[(size_t)row * 2048 + c2] = bv;
          } else if (sel == 1) {
            o1[(size_t)row * 2048 + c2] = bv;
          } else {  // V stored pre-transposed per head: [b][h][hd][seq]
            const int bb = row >> 10, sq = row & 1023;
            const int hh = c2 >> 7, e = c2 & 127;
            o2[((size_t)(bb * 16 + hh) * 128 + e) * 1024 + sq] = bv;
          }
        } else if constexpr (MODE == 1) {
          outF[(size_t)row * N + col] = resid[(size_t)row * N + col] + v;
        } else {  // exact GELU
          const float gl = 0.5f * v * (1.f + erff(v * 0.70710678118654752f));
          o0[(size_t)row * N + col] = f2bf(gl);
        }
      }
    }
  }
}

// ---------------------------------------------------------------------------
// Flash attention with ALiBi + causal mask.
// grid (seq/128, H, batch); 8 waves, each owns 16 q rows.
// S = Q*K^T via WMMA from global (K row-major along hd, L2-resident);
// P transposed through per-wave LDS; P*V via WMMA against pre-transposed V.
// No block barriers inside the (wave-divergent) kv loop.
// ---------------------------------------------------------------------------
__global__ __launch_bounds__(256) void attn_k(
    const u16* __restrict__ Q, const u16* __restrict__ Kb,
    const u16* __restrict__ Vt, const float* __restrict__ slopes,
    u16* __restrict__ ctx) {
  __shared__ alignas(16) u16 sP[8][16][64];   // per-wave P tile (16 q x 64 kv)
  const int tid = threadIdx.x, lane = tid & 31, w = tid >> 5;
  const int g = lane >> 4, lm = lane & 15;
  const int qblk = blockIdx.x * 128, h = blockIdx.y, b = blockIdx.z;
  const int q0 = qblk + w * 16;

  // Q fragments (A layout), loaded once: row m = lm, 4 chunks of K(=hd) 32
  FragU fq[4];
  {
    const size_t qoff = ((size_t)(b * 1024 + q0 + lm)) * 2048 + h * 128;
#pragma unroll
    for (int ec = 0; ec < 4; ++ec) {
      fq[ec].u[0] = *(const uint4*)(Q + qoff + ec * 32 + 16 * g);
      fq[ec].u[1] = *(const uint4*)(Q + qoff + ec * 32 + 16 * g + 8);
    }
  }

  f32x8 o[8];
#pragma unroll
  for (int et = 0; et < 8; ++et) o[et] = zero8();
  float rmax[8], rsum[8];
#pragma unroll
  for (int r = 0; r < 8; ++r) { rmax[r] = -3e38f; rsum[r] = 0.f; }

  const float slope = slopes[h];
  const float sc = 0.088388347648318447f;  // 1/sqrt(128)
  const size_t kbase = ((size_t)(b * 1024)) * 2048 + h * 128;
  const size_t vbase = ((size_t)(b * 16 + h)) * 128 * 1024;

  for (int kv = 0; kv < q0 + 16; kv += 64) {      // causal upper bound
    // ---- S = Q K^T (4 n-tiles of 16 kv tokens, K-dim = hd = 128) ----
    f32x8 sacc[4];
#pragma unroll
    for (int nt = 0; nt < 4; ++nt) {
      sacc[nt] = zero8();
      const size_t krow = kbase + (size_t)(kv + nt * 16 + lm) * 2048;
#pragma unroll
      for (int ec = 0; ec < 4; ++ec) {
        FragU fk;
        fk.u[0] = *(const uint4*)(Kb + krow + ec * 32 + 16 * g);
        fk.u[1] = *(const uint4*)(Kb + krow + ec * 32 + 16 * g + 8);
        sacc[nt] = __builtin_amdgcn_wmma_f32_16x16x32_bf16(
            false, fq[ec].v, false, fk.v, (short)0, sacc[nt], false, false);
      }
    }
    // ---- online softmax (C layout: col = lm per n-tile, row = r + 8g) ----
#pragma unroll
    for (int r = 0; r < 8; ++r) {
      const int q = q0 + 8 * g + r;
      float pv[4];
      float cmax = -3e38f;
#pragma unroll
      for (int nt = 0; nt < 4; ++nt) {
        const int k = kv + nt * 16 + lm;
        float val = sacc[nt][r] * sc + slope * (float)(k - q);
        if (k > q) val = -3e38f;
        pv[nt] = val;
        cmax = fmaxf(cmax, val);
      }
#pragma unroll
      for (int m = 1; m < 16; m <<= 1) cmax = fmaxf(cmax, __shfl_xor(cmax, m, 32));
      const float nm = fmaxf(rmax[r], cmax);
      const float corr = __expf(rmax[r] - nm);
      rmax[r] = nm;
      float psum = 0.f;
#pragma unroll
      for (int nt = 0; nt < 4; ++nt) {
        const float p = __expf(pv[nt] - nm);
        pv[nt] = p;
        psum += p;
      }
#pragma unroll
      for (int m = 1; m < 16; m <<= 1) psum += __shfl_xor(psum, m, 32);
      rsum[r] = rsum[r] * corr + psum;
#pragma unroll
      for (int et = 0; et < 8; ++et) o[et][r] *= corr;
#pragma unroll
      for (int nt = 0; nt < 4; ++nt)
        sP[w][8 * g + r][nt * 16 + lm] = f2bf(pv[nt]);
    }
    asm volatile("s_wait_dscnt 0x0" ::: "memory");  // P stores -> P loads (same wave)
    // ---- O += P * V  (K-dim = 64 kv, two wmma steps; V pre-transposed) ----
#pragma unroll
    for (int kc = 0; kc < 2; ++kc) {
      FragU fp;
      fp.u[0] = *(const uint4*)&sP[w][lm][kc * 32 + 16 * g];
      fp.u[1] = *(const uint4*)&sP[w][lm][kc * 32 + 16 * g + 8];
#pragma unroll
      for (int et = 0; et < 8; ++et) {
        const size_t vrow =
            vbase + (size_t)(et * 16 + lm) * 1024 + kv + kc * 32 + 16 * g;
        FragU fv;
        fv.u[0] = *(const uint4*)(Vt + vrow);
        fv.u[1] = *(const uint4*)(Vt + vrow + 8);
        o[et] = __builtin_amdgcn_wmma_f32_16x16x32_bf16(
            false, fp.v, false, fv.v, (short)0, o[et], false, false);
      }
    }
  }
  // ---- normalize + store ctx (bf16, [T][2048]) ----
#pragma unroll
  for (int r = 0; r < 8; ++r) {
    const float inv = 1.f / rsum[r];
    const size_t row = ((size_t)(b * 1024 + q0 + 8 * g + r)) * 2048 + h * 128;
#pragma unroll
    for (int et = 0; et < 8; ++et)
      ctx[row + et * 16 + lm] = f2bf(o[et][r] * inv);
  }
}

// ---------------------------------------------------------------------------
// Host-side orchestration
// ---------------------------------------------------------------------------
extern "C" void kernel_launch(void* const* d_in, const int* in_sizes, int n_in,
                              void* d_out, int out_size, void* d_ws,
                              size_t ws_size, hipStream_t stream) {
  (void)in_sizes; (void)n_in; (void)out_size; (void)ws_size;
  const float* hidden = (const float*)d_in[0];
  const float* w1     = (const float*)d_in[1];
  const float* w2     = (const float*)d_in[2];
  const float* Wqkv   = (const float*)d_in[3];
  const float* Wout   = (const float*)d_in[4];
  const float* Wup    = (const float*)d_in[5];
  const float* Wdown  = (const float*)d_in[6];
  const float* slopes = (const float*)d_in[7];
  float* out = (float*)d_out;

  constexpr size_t MB = 1ull << 20;
  char* ws = (char*)d_ws;
  u16*   WT   = (u16*)(ws + 0);          // weight^T scratch (up to 40MB span)
  u16*   X1   = (u16*)(ws + 24 * MB);    // ln1 out; later ctx (16MB)
  u16*   Qb   = (u16*)(ws + 40 * MB);    // 16MB
  u16*   Kbuf = (u16*)(ws + 56 * MB);    // 16MB
  u16*   Vtb  = (u16*)(ws + 72 * MB);    // 16MB
  float* Hbuf = (float*)(ws + 40 * MB);  // 32MB (reuses Q+K after attention)
  u16*   X2   = (u16*)(ws + 72 * MB);    // reuses Vt
  u16*   UP   = (u16*)(ws + 88 * MB);    // 64MB

  const dim3 blk(256);

  // 1) Wqkv^T (bf16 [6144][2048])
  transpose_f32_bf16<<<dim3(6144 / 32, 2048 / 32), blk, 0, stream>>>(Wqkv, WT, 2048, 6144);
  // 2) x1 = LN(hidden, w1)
  layernorm_k<<<dim3(4096), blk, 0, stream>>>(hidden, w1, X1);
  // 3) QKV GEMM, split epilogue (V stored per-head transposed)
  gemm_bf16<0><<<dim3(6144 / 128, 4096 / 128), blk, 0, stream>>>(
      X1, WT, 4096, 6144, 2048, nullptr, nullptr, Qb, Kbuf, Vtb);
  // 4) attention -> ctx (overwrites X1)
  attn_k<<<dim3(1024 / 128, 16, 4), blk, 0, stream>>>(Qb, Kbuf, Vtb, slopes, X1);
  // 5) Wout^T
  transpose_f32_bf16<<<dim3(2048 / 32, 2048 / 32), blk, 0, stream>>>(Wout, WT, 2048, 2048);
  // 6) h = hidden + ctx @ Wout
  gemm_bf16<1><<<dim3(2048 / 128, 4096 / 128), blk, 0, stream>>>(
      X1, WT, 4096, 2048, 2048, hidden, Hbuf, nullptr, nullptr, nullptr);
  // 7) x2 = LN(h, w2)
  layernorm_k<<<dim3(4096), blk, 0, stream>>>(Hbuf, w2, X2);
  // 8) Wup^T (32MB, spans WT+X1 region; ctx dead by now)
  transpose_f32_bf16<<<dim3(8192 / 32, 2048 / 32), blk, 0, stream>>>(Wup, (u16*)ws, 2048, 8192);
  // 9) up = gelu(x2 @ Wup) (bf16)
  gemm_bf16<2><<<dim3(8192 / 128, 4096 / 128), blk, 0, stream>>>(
      X2, (u16*)ws, 4096, 8192, 2048, nullptr, nullptr, UP, nullptr, nullptr);
  // 10) Wdown^T (overwrites Wup^T)
  transpose_f32_bf16<<<dim3(2048 / 32, 8192 / 32), blk, 0, stream>>>(Wdown, (u16*)ws, 8192, 2048);
  // 11) out = h + up @ Wdown
  gemm_bf16<1><<<dim3(2048 / 128, 4096 / 128), blk, 0, stream>>>(
      UP, (u16*)ws, 4096, 2048, 8192, Hbuf, out, nullptr, nullptr, nullptr);
}
